// DSSM_57629871178390
// MI455X (gfx1250) — matrix operand order repaired and checked by reading, not compiled, not run on hardware
//
#include <hip/hip_runtime.h>
#include <hip/hip_bf16.h>
#include <math.h>

// ---------------------------------------------------------------------------
// DSSM (dual-stream Mamba selective scan) for MI455X / gfx1250
//   - all GEMMs via v_wmma_f32_16x16x32_bf16 (fp32 accumulate)
//   - wave32-aware scan: 2 channels x 16 states per wave, shfl_xor reductions
//   - global_atomic_add_f32 merge of K=3 scan directions
//   - global_prefetch_b8 on the sequential scan's critical path
// ---------------------------------------------------------------------------

typedef __bf16 bf16;
typedef __bf16 v16bf __attribute__((ext_vector_type(16)));
typedef float  v8f   __attribute__((ext_vector_type(8)));

#define BM  128
#define BN  64
#define BKT 32
#define LDA 34   // BKT + 2 padding: breaks LDS bank conflicts

static __device__ __forceinline__ v8f v8f_zero() {
    v8f z;
#pragma unroll
    for (int e = 0; e < 8; ++e) z[e] = 0.f;
    return z;
}

static __device__ __forceinline__ v8f wmma_bf16(v16bf a, v16bf b, v8f c) {
    // D = A(16x32 bf16) * B(32x16 bf16) + C(16x16 f32)
    return __builtin_amdgcn_wmma_f32_16x16x32_bf16(false, a, false, b, (short)0, c,
                                                   false, false);
}

// A fragment: 16x32 bf16.  lanes 0-15: rows M=0..15, K 0-7 / 16-23;
// lanes 16-31: same rows, K 8-15 / 24-31 (ISA 7.12.2 layout).
static __device__ __forceinline__ v16bf load_a_frag(const bf16* As, int rowBase, int lane) {
    const int l15 = lane & 15;
    const int b0  = (lane < 16) ? 0 : 8;
    const int b1  = (lane < 16) ? 16 : 24;
    const bf16* p = As + (rowBase + l15) * LDA;
    v16bf a;
#pragma unroll
    for (int e = 0; e < 8; ++e) a[e] = p[b0 + e];
#pragma unroll
    for (int e = 0; e < 8; ++e) a[8 + e] = p[b1 + e];
    return a;
}

// B fragment: 32x16 bf16. LDS holds B transposed as [n][k]; each lane reads
// K values for its column n; lanes 0-15 K 0-7 / 16-23, 16-31 K 8-15 / 24-31.
static __device__ __forceinline__ v16bf load_b_frag(const bf16* Bs, int colBase, int lane) {
    const int l15 = lane & 15;
    const int b0  = (lane < 16) ? 0 : 8;
    const int b1  = (lane < 16) ? 16 : 24;
    const bf16* p = Bs + (colBase + l15) * LDA;
    v16bf b;
#pragma unroll
    for (int e = 0; e < 8; ++e) b[e] = p[b0 + e];
#pragma unroll
    for (int e = 0; e < 8; ++e) b[8 + e] = p[b1 + e];
    return b;
}

// Stage B tile from an (Nsrc x Kld) fp32 row-major weight (contiguous in K),
// converting to bf16; rows >= Nvalid are zero-padded.
static __device__ __forceinline__ void stage_b(bf16* Bs, const float* __restrict__ Wsrc,
                                               int k0, int Nvalid, int Kld, int tid) {
#pragma unroll
    for (int j = 0; j < (BN * BKT) / 256; ++j) {
        int idx = j * 256 + tid;
        int nr  = idx >> 5;   // 0..63
        int kc  = idx & 31;
        float v = (nr < Nvalid) ? Wsrc[(size_t)nr * Kld + k0 + kc] : 0.f;
        Bs[nr * LDA + kc] = (bf16)v;
    }
}

// ---------------------------------------------------------------------------
// Kernel 1: in_proj GEMM  Y[m,n] = sum_c src[b,c,l] * W[n,c]
//   M = 8192 tokens (b-major, l within), K = 128, N = 512
//   n <  256 -> x part, written NCHW (b,d,l) for the conv
//   n >= 256 -> z part, written token-major (b,l,d) for gating
// ---------------------------------------------------------------------------
__global__ __launch_bounds__(256) void k_inproj(
    const float* __restrict__ ms, const float* __restrict__ pan,
    const float* __restrict__ w_ms, const float* __restrict__ w_pan,
    float* __restrict__ x_ms, float* __restrict__ x_pan,
    float* __restrict__ z_ms, float* __restrict__ z_pan)
{
    __shared__ bf16 As[BM * LDA];
    __shared__ bf16 Bs[BN * LDA];
    const int tid  = threadIdx.x;
    const int lane = tid & 31;
    const int wid  = tid >> 5;
    const int wm   = wid & 3;
    const int wn   = wid >> 2;
    const int s    = blockIdx.z;
    const float* src  = s ? pan  : ms;
    const float* Wsrc = s ? w_pan : w_ms;
    float* xb = s ? x_pan : x_ms;
    float* zb = s ? z_pan : z_ms;
    const int n0 = blockIdx.x * BN;
    const int m0 = blockIdx.y * BM;
    const int b  = m0 >> 12;
    const int l0 = m0 & 4095;

    v8f acc[2][2];
#pragma unroll
    for (int i = 0; i < 2; ++i)
#pragma unroll
        for (int j = 0; j < 2; ++j) acc[i][j] = v8f_zero();

    for (int k0 = 0; k0 < 128; k0 += BKT) {
        // stage A: A[m][c] = src[b, k0+c, l0+m]; coalesced along l
#pragma unroll
        for (int j = 0; j < 16; ++j) {
            int idx = j * 256 + tid;
            int c   = idx >> 7;    // 0..31
            int r   = idx & 127;   // 0..127 (token row)
            As[r * LDA + c] = (bf16)src[((size_t)(b * 128 + k0 + c)) * 4096 + (l0 + r)];
        }
        stage_b(Bs, Wsrc, k0, 512, 128, tid);
        __syncthreads();
        v16bf a0 = load_a_frag(As, wm * 32 + 0,  lane);
        v16bf a1 = load_a_frag(As, wm * 32 + 16, lane);
        v16bf b0 = load_b_frag(Bs, wn * 32 + 0,  lane);
        v16bf b1 = load_b_frag(Bs, wn * 32 + 16, lane);
        acc[0][0] = wmma_bf16(a0, b0, acc[0][0]);
        acc[0][1] = wmma_bf16(a0, b1, acc[0][1]);
        acc[1][0] = wmma_bf16(a1, b0, acc[1][0]);
        acc[1][1] = wmma_bf16(a1, b1, acc[1][1]);
        __syncthreads();
    }

    const int l15   = lane & 15;
    const int hiOff = (lane >= 16) ? 8 : 0;
#pragma unroll
    for (int i = 0; i < 2; ++i)
#pragma unroll
        for (int j = 0; j < 2; ++j) {
            int gn = n0 + wn * 32 + j * 16 + l15;
#pragma unroll
            for (int r = 0; r < 8; ++r) {
                int gm = m0 + wm * 32 + i * 16 + r + hiOff;
                int l  = gm & 4095;
                int bb = gm >> 12;
                float v = acc[i][j][r];
                if (gn < 256)
                    xb[((size_t)(bb * 256 + gn)) * 4096 + l] = v;
                else
                    zb[((size_t)(bb * 4096) + l) * 256 + (gn - 256)] = v;
            }
        }
}

// ---------------------------------------------------------------------------
// Kernel 2: depthwise 3x3 SAME conv + SiLU; writes interleaved scan layout
//   xst[b, t=2l+s, d]
// ---------------------------------------------------------------------------
__global__ __launch_bounds__(256) void k_dwconv(
    const float* __restrict__ x_ms, const float* __restrict__ x_pan,
    const float* __restrict__ cw_ms, const float* __restrict__ cb_ms,
    const float* __restrict__ cw_pan, const float* __restrict__ cb_pan,
    float* __restrict__ xst)
{
    const int s   = blockIdx.y;
    const int idx = blockIdx.x * 256 + threadIdx.x;   // 0 .. 2*256*4096-1
    const int b   = idx >> 20;
    const int rem = idx & 1048575;
    const int d   = rem >> 12;
    const int l   = rem & 4095;
    const int h   = l >> 6;
    const int w   = l & 63;
    const float* src = (s ? x_pan : x_ms) + ((size_t)(b * 256 + d)) * 4096;
    const float* wc  = (s ? cw_pan : cw_ms) + d * 9;
    float acc = (s ? cb_pan : cb_ms)[d];
#pragma unroll
    for (int dy = 0; dy < 3; ++dy) {
        int hh = h + dy - 1;
        if (hh < 0 || hh > 63) continue;
#pragma unroll
        for (int dx = 0; dx < 3; ++dx) {
            int ww = w + dx - 1;
            if (ww < 0 || ww > 63) continue;
            acc += src[hh * 64 + ww] * wc[dy * 3 + dx];
        }
    }
    float y = acc / (1.f + __expf(-acc));            // SiLU
    xst[((size_t)(b * 8192) + 2 * l + s) * 256 + d] = y;
}

// ---------------------------------------------------------------------------
// Kernel 3: x_dbl GEMM  xdbl[b,k,t,c] = sum_d xs[b,k,d,t] * Wk[c,d]
//   per (b,k): M = 8192, K = 256, N = 40 (padded to 48 in storage, 64 in tile)
//   k==1 direction flipped during A staging.
// ---------------------------------------------------------------------------
__global__ __launch_bounds__(256) void k_xdbl(
    const float* __restrict__ xst, const float* __restrict__ xproj_w,
    float* __restrict__ xdbl)
{
    __shared__ bf16 As[BM * LDA];
    __shared__ bf16 Bs[BN * LDA];
    const int tid  = threadIdx.x;
    const int lane = tid & 31;
    const int wid  = tid >> 5;
    const int wm   = wid & 3;
    const int wn   = wid >> 2;
    const int z    = blockIdx.z;      // b*3 + k
    const int b    = z / 3;
    const int k    = z % 3;
    const int m0   = blockIdx.y * BM;
    const float* Wsrc = xproj_w + (size_t)k * 40 * 256;

    v8f acc[2][2];
#pragma unroll
    for (int i = 0; i < 2; ++i)
#pragma unroll
        for (int j = 0; j < 2; ++j) acc[i][j] = v8f_zero();

    for (int k0 = 0; k0 < 256; k0 += BKT) {
#pragma unroll
        for (int j = 0; j < 16; ++j) {
            int idx = j * 256 + tid;
            int r   = idx >> 5;
            int c   = idx & 31;
            int t   = m0 + r;
            int tt  = (k == 1) ? (8191 - t) : t;
            As[r * LDA + c] = (bf16)xst[((size_t)(b * 8192) + tt) * 256 + k0 + c];
        }
        stage_b(Bs, Wsrc, k0, 40, 256, tid);
        __syncthreads();
        v16bf a0 = load_a_frag(As, wm * 32 + 0,  lane);
        v16bf a1 = load_a_frag(As, wm * 32 + 16, lane);
        v16bf b0 = load_b_frag(Bs, wn * 32 + 0,  lane);
        v16bf b1 = load_b_frag(Bs, wn * 32 + 16, lane);
        acc[0][0] = wmma_bf16(a0, b0, acc[0][0]);
        acc[0][1] = wmma_bf16(a0, b1, acc[0][1]);
        acc[1][0] = wmma_bf16(a1, b0, acc[1][0]);
        acc[1][1] = wmma_bf16(a1, b1, acc[1][1]);
        __syncthreads();
    }

    const int l15   = lane & 15;
    const int hiOff = (lane >= 16) ? 8 : 0;
#pragma unroll
    for (int i = 0; i < 2; ++i)
#pragma unroll
        for (int j = 0; j < 2; ++j) {
            int gn = wn * 32 + j * 16 + l15;
            if (gn >= 48) continue;
#pragma unroll
            for (int r = 0; r < 8; ++r) {
                int gm = m0 + wm * 32 + i * 16 + r + hiOff;   // scan position t
                xdbl[((size_t)z * 8192 + gm) * 48 + gn] = acc[i][j][r];
            }
        }
}

// ---------------------------------------------------------------------------
// Kernel 4: selective scan. One block = (b, k, 32-channel chunk); 512 threads.
//   thread = (channel_local, state n): each wave holds 2 channels x 16 states.
//   Fuses dt-projection + softplus; 16-lane shfl_xor reduction for y;
//   merges K directions with global f32 atomics (k=1 un-flipped via tt).
// ---------------------------------------------------------------------------
__global__ __launch_bounds__(512) void k_scan(
    const float* __restrict__ xdbl, const float* __restrict__ xst,
    const float* __restrict__ dtw_all, const float* __restrict__ dtb_all,
    const float* __restrict__ A_logs, const float* __restrict__ Ds,
    float* __restrict__ ybuf)
{
    const int blk   = blockIdx.x;     // 0..47
    const int chunk = blk & 7;        // 8 chunks of 32 channels
    const int zk    = blk >> 3;       // b*3 + k
    const int b     = zk / 3;
    const int k     = zk % 3;
    const int tid   = threadIdx.x;
    const int chl   = tid >> 4;       // 0..31
    const int n     = tid & 15;       // state index
    const int d     = chunk * 32 + chl;
    const int kd    = k * 256 + d;

    const float Aval = -__expf(A_logs[kd * 16 + n]);
    const float Dval = Ds[kd];
    const float dtb  = dtb_all[kd];
    float dtw[8];
#pragma unroll
    for (int r = 0; r < 8; ++r) dtw[r] = dtw_all[kd * 8 + r];

    const float* xdrow = xdbl + (size_t)zk * 8192 * 48;
    float h = 0.f;

    for (int t = 0; t < 8192; ++t) {
        const float* xd = xdrow + (size_t)t * 48;
        if (t + 1 < 8192) __builtin_prefetch(xd + 48, 0, 3);   // global_prefetch_b8

        float dtlin = dtb;
#pragma unroll
        for (int r = 0; r < 8; ++r) dtlin += xd[r] * dtw[r];
        float dt = (dtlin > 20.f) ? dtlin : __logf(1.f + __expf(dtlin));  // softplus

        float Bn = xd[8 + n];
        float Cn = xd[24 + n];
        int tt = (k == 1) ? (8191 - t) : t;
        float x = xst[((size_t)(b * 8192) + tt) * 256 + d];

        h = __expf(dt * Aval) * h + (dt * x) * Bn;

        float c = h * Cn;
#pragma unroll
        for (int off = 8; off > 0; off >>= 1) c += __shfl_xor(c, off, 16);
        if (n == 0)
            atomicAdd(&ybuf[((size_t)(b * 8192) + tt) * 256 + d], c + Dval * x);
    }
}

// ---------------------------------------------------------------------------
// Kernel 5: per-token LayerNorm over d=256 + SiLU(z) gate -> bf16 activations
// ---------------------------------------------------------------------------
__global__ __launch_bounds__(256) void k_ln_gate(
    const float* __restrict__ ybuf,
    const float* __restrict__ z_ms, const float* __restrict__ z_pan,
    const float* __restrict__ vis_w, const float* __restrict__ vis_b,
    const float* __restrict__ inf_w, const float* __restrict__ inf_b,
    bf16* __restrict__ g_ms, bf16* __restrict__ g_pan)
{
    __shared__ float red[8];
    __shared__ float bc[2];
    const int bid = blockIdx.x;
    const int s   = bid >> 13;
    const int tk  = bid & 8191;
    const int b   = tk >> 12;
    const int l   = tk & 4095;
    const int t   = 2 * l + s;
    const int d   = threadIdx.x;
    const int lane = d & 31, wid = d >> 5;

    float v = ybuf[((size_t)(b * 8192) + t) * 256 + d];

    float sum = v;
#pragma unroll
    for (int off = 16; off > 0; off >>= 1) sum += __shfl_xor(sum, off, 32);
    if (lane == 0) red[wid] = sum;
    __syncthreads();
    if (d == 0) {
        float tot = 0.f;
        for (int i = 0; i < 8; ++i) tot += red[i];
        bc[0] = tot * (1.f / 256.f);
    }
    __syncthreads();
    const float mu = bc[0];
    const float xc = v - mu;

    float sq = xc * xc;
#pragma unroll
    for (int off = 16; off > 0; off >>= 1) sq += __shfl_xor(sq, off, 32);
    if (lane == 0) red[wid] = sq;
    __syncthreads();
    if (d == 0) {
        float tot = 0.f;
        for (int i = 0; i < 8; ++i) tot += red[i];
        bc[1] = tot * (1.f / 256.f);
    }
    __syncthreads();
    const float rs = rsqrtf(bc[1] + 1e-5f);

    const float wv = (s ? inf_w : vis_w)[d];
    const float bv = (s ? inf_b : vis_b)[d];
    const float zv = (s ? z_pan : z_ms)[((size_t)(b * 4096) + l) * 256 + d];
    const float g  = (xc * rs * wv + bv) * (zv / (1.f + __expf(-zv)));
    (s ? g_pan : g_ms)[((size_t)(b * 4096) + l) * 256 + d] = (bf16)g;
}

// ---------------------------------------------------------------------------
// Kernel 6: out_proj GEMM  out[b,o,l] = sum_d g[b,l,d] * W[o,d]
//   M = 8192, K = 256, N = 128; writes straight into d_out (ms then pan).
// ---------------------------------------------------------------------------
__global__ __launch_bounds__(256) void k_outproj(
    const bf16* __restrict__ g_ms, const bf16* __restrict__ g_pan,
    const float* __restrict__ w_ms, const float* __restrict__ w_pan,
    float* __restrict__ out)
{
    __shared__ bf16 As[BM * LDA];
    __shared__ bf16 Bs[BN * LDA];
    const int tid  = threadIdx.x;
    const int lane = tid & 31;
    const int wid  = tid >> 5;
    const int wm   = wid & 3;
    const int wn   = wid >> 2;
    const int s    = blockIdx.z;
    const bf16*  A    = s ? g_pan : g_ms;
    const float* Wsrc = s ? w_pan : w_ms;
    const int n0 = blockIdx.x * BN;
    const int m0 = blockIdx.y * BM;

    v8f acc[2][2];
#pragma unroll
    for (int i = 0; i < 2; ++i)
#pragma unroll
        for (int j = 0; j < 2; ++j) acc[i][j] = v8f_zero();

    for (int k0 = 0; k0 < 256; k0 += BKT) {
#pragma unroll
        for (int j = 0; j < 16; ++j) {
            int idx = j * 256 + tid;
            int r   = idx >> 5;
            int c   = idx & 31;
            As[r * LDA + c] = A[(size_t)(m0 + r) * 256 + k0 + c];
        }
        stage_b(Bs, Wsrc, k0, 128, 256, tid);
        __syncthreads();
        v16bf a0 = load_a_frag(As, wm * 32 + 0,  lane);
        v16bf a1 = load_a_frag(As, wm * 32 + 16, lane);
        v16bf b0 = load_b_frag(Bs, wn * 32 + 0,  lane);
        v16bf b1 = load_b_frag(Bs, wn * 32 + 16, lane);
        acc[0][0] = wmma_bf16(a0, b0, acc[0][0]);
        acc[0][1] = wmma_bf16(a0, b1, acc[0][1]);
        acc[1][0] = wmma_bf16(a1, b0, acc[1][0]);
        acc[1][1] = wmma_bf16(a1, b1, acc[1][1]);
        __syncthreads();
    }

    const int l15   = lane & 15;
    const int hiOff = (lane >= 16) ? 8 : 0;
#pragma unroll
    for (int i = 0; i < 2; ++i)
#pragma unroll
        for (int j = 0; j < 2; ++j) {
            int gn = n0 + wn * 32 + j * 16 + l15;   // output channel o (<128)
#pragma unroll
            for (int r = 0; r < 8; ++r) {
                int gm = m0 + wm * 32 + i * 16 + r + hiOff;
                int l  = gm & 4095;
                int bb = gm >> 12;
                out[(size_t)s * 1048576 + ((size_t)(bb * 128 + gn)) * 4096 + l] =
                    acc[i][j][r];
            }
        }
}

// ---------------------------------------------------------------------------
extern "C" void kernel_launch(void* const* d_in, const int* in_sizes, int n_in,
                              void* d_out, int out_size, void* d_ws, size_t ws_size,
                              hipStream_t stream) {
    (void)in_sizes; (void)n_in; (void)out_size; (void)ws_size;

    const float* ms        = (const float*)d_in[0];
    const float* pan       = (const float*)d_in[1];
    const float* w_in_ms   = (const float*)d_in[2];   // (512,128)
    const float* w_in_pan  = (const float*)d_in[3];
    const float* cw_ms     = (const float*)d_in[4];   // (256,1,3,3)
    const float* cb_ms     = (const float*)d_in[5];
    const float* cw_pan    = (const float*)d_in[6];
    const float* cb_pan    = (const float*)d_in[7];
    const float* xproj_w   = (const float*)d_in[8];   // (3,40,256)
    const float* dtw       = (const float*)d_in[9];   // (3,256,8)
    const float* dtb       = (const float*)d_in[10];  // (3,256)
    const float* A_logs    = (const float*)d_in[11];  // (768,16)
    const float* Ds        = (const float*)d_in[12];  // (768)
    const float* vis_w     = (const float*)d_in[13];
    const float* vis_b     = (const float*)d_in[14];
    const float* inf_w     = (const float*)d_in[15];
    const float* inf_b     = (const float*)d_in[16];
    const float* wout_ms   = (const float*)d_in[17];  // (128,256)
    const float* wout_pan  = (const float*)d_in[18];

    char* ws = (char*)d_ws;
    size_t off = 0;
    auto take = [&](size_t bytes) -> void* {
        void* p = (void*)(ws + off);
        off += (bytes + 255) & ~(size_t)255;
        return p;
    };

    float* x_ms  = (float*)take(8388608);    // (B,256,L) conv input, ms
    float* x_pan = (float*)take(8388608);
    float* z_ms  = (float*)take(8388608);    // (B,L,256) gate, ms
    float* z_pan = (float*)take(8388608);
    float* xst   = (float*)take(16777216);   // (B,2L,256) interleaved scan input
    float* xdbl  = (float*)take(9437184);    // (B,3,2L,48) dt/B/C rows
    float* ybuf  = (float*)take(16777216);   // (B,2L,256) merged scan output
    bf16*  g_ms  = (bf16*)take(4194304);     // (B,L,256) bf16 gated activations
    bf16*  g_pan = (bf16*)take(4194304);

    hipMemsetAsync(ybuf, 0, 16777216, stream);

    k_inproj<<<dim3(8, 64, 2), 256, 0, stream>>>(ms, pan, w_in_ms, w_in_pan,
                                                 x_ms, x_pan, z_ms, z_pan);
    k_dwconv<<<dim3(8192, 2), 256, 0, stream>>>(x_ms, x_pan, cw_ms, cb_ms,
                                                cw_pan, cb_pan, xst);
    k_xdbl<<<dim3(1, 64, 6), 256, 0, stream>>>(xst, xproj_w, xdbl);
    k_scan<<<dim3(48), 512, 0, stream>>>(xdbl, xst, dtw, dtb, A_logs, Ds, ybuf);
    k_ln_gate<<<dim3(16384), 256, 0, stream>>>(ybuf, z_ms, z_pan,
                                               vis_w, vis_b, inf_w, inf_b,
                                               g_ms, g_pan);
    k_outproj<<<dim3(2, 64, 2), 256, 0, stream>>>(g_ms, g_pan, wout_ms, wout_pan,
                                                  (float*)d_out);
}
